// NeuralComponent_4432406250089
// MI455X (gfx1250) — compile-verified, matrix operand order checked by reading, not run on hardware
//
#include <hip/hip_runtime.h>
#include <hip/hip_bf16.h>

// ---------------------------------------------------------------------------
// MI455X / gfx1250 implementation: bf16 WMMA for every GEMM, persistent
// single-WGP LSTM scan with the whole recurrent state in LDS, wave-fused LN.
// ---------------------------------------------------------------------------

typedef __attribute__((ext_vector_type(16))) __bf16 v16bf;
typedef __attribute__((ext_vector_type(8)))  __bf16 v8bf;
typedef __attribute__((ext_vector_type(8)))  float  v8f;

#define B_   16
#define S_   512
#define DI_  64
#define H_   128
#define DO_  64
#define L_   3
#define M_   32
#define BS_  (B_ * S_)          // 8192 rows
#define H3_  (3 * H_)           // 384

__device__ __forceinline__ unsigned short f2bf(float f) {
  union { float f; unsigned u; } x; x.f = f;
  unsigned r = x.u + 0x7FFFu + ((x.u >> 16) & 1u);   // RNE
  return (unsigned short)(r >> 16);
}
__device__ __forceinline__ float bf2f(unsigned short h) {
  union { unsigned u; float f; } x; x.u = ((unsigned)h) << 16; return x.f;
}
__device__ __forceinline__ float sigm(float x) { return 1.f / (1.f + __expf(-x)); }
__device__ __forceinline__ v8f vzero8() {
  v8f z = {0.f,0.f,0.f,0.f,0.f,0.f,0.f,0.f}; return z;
}

// A (16xK) or B(=W row-major [n][k]) fragment from a row-major bf16 buffer.
// ISA 16-bit A layout: lanes 0-15 hold K runs {k0..k0+7, k0+16..k0+23},
// lanes 16-31 hold {k0+8..k0+15, k0+24..k0+31}; row = (lane&15)-indexed.
__device__ __forceinline__ v16bf frag_rm(const unsigned short* base, int stride,
                                         int row, int k0, int lane) {
  int kh = (lane & 16) ? 8 : 0;
  const unsigned short* p = base + row * stride + k0 + kh;
  v8bf lo = *reinterpret_cast<const v8bf*>(p);
  v8bf hi = *reinterpret_cast<const v8bf*>(p + 16);
  return __builtin_shufflevector(lo, hi, 0,1,2,3,4,5,6,7,8,9,10,11,12,13,14,15);
}

// B fragment from a k-major buffer: element (k, n) = base[k*stride + n].
__device__ __forceinline__ v16bf frag_km(const unsigned short* base, int stride,
                                         int n, int k0, int lane) {
  int kh = (lane & 16) ? 8 : 0;
  const __bf16* b = reinterpret_cast<const __bf16*>(base);
  v16bf f;
#pragma unroll
  for (int i = 0; i < 8; ++i) f[i]     = b[(k0 + kh + i) * stride + n];
#pragma unroll
  for (int i = 0; i < 8; ++i) f[i + 8] = b[(k0 + 16 + kh + i) * stride + n];
  return f;
}

__device__ __forceinline__ v8f wmma_bf(v16bf a, v16bf b, v8f c) {
  return __builtin_amdgcn_wmma_f32_16x16x32_bf16(false, a, false, b,
                                                 (short)0, c, false, false);
}

// ---------------------------------------------------------------------------
__global__ void k_f2bf(const float* __restrict__ src,
                       unsigned short* __restrict__ dst, int n) {
  int i = blockIdx.x * blockDim.x + threadIdx.x;
  if (i < n) dst[i] = f2bf(src[i]);
}

// ---------------------------------------------------------------------------
// Kernel 1: input proj + 3 residual MLP+LN layers (wave-fused LN).
// ---------------------------------------------------------------------------
__global__ __launch_bounds__(256)
void k_mlp(const float* __restrict__ x, const unsigned short* __restrict__ Win_bf,
           const float* __restrict__ b_in, const unsigned short* __restrict__ Wl_bf,
           const float* __restrict__ b_l, const float* __restrict__ g_l,
           const float* __restrict__ be_l, unsigned short* __restrict__ h_bf) {
  __shared__ unsigned short strip[8][16][H_];
  int lane = threadIdx.x & 31;
  int w    = threadIdx.x >> 5;
  int r0   = blockIdx.x * 128 + w * 16;
  int am   = lane & 15;
  int rb   = (lane & 16) ? 8 : 0;
  unsigned short (*A)[H_] = strip[w];

  for (int j = lane; j < 16 * DI_; j += 32) {
    int r = j >> 6, c = j & 63;
    A[r][c] = f2bf(x[(size_t)(r0 + r) * DI_ + c]);
  }
  __syncthreads();

  v8f acc[8];
#pragma unroll
  for (int t = 0; t < 8; ++t) acc[t] = vzero8();
  for (int k0 = 0; k0 < DI_; k0 += 32) {
    v16bf a = frag_rm(&A[0][0], H_, am, k0, lane);
#pragma unroll
    for (int t = 0; t < 8; ++t) {
      v16bf b = frag_rm(Win_bf, DI_, t * 16 + am, k0, lane);
      acc[t] = wmma_bf(a, b, acc[t]);
    }
  }
#pragma unroll
  for (int t = 0; t < 8; ++t) {
    int col = t * 16 + am;
    float bias = b_in[col];
#pragma unroll
    for (int r = 0; r < 8; ++r) A[rb + r][col] = f2bf(acc[t][r] + bias);
  }
  __syncthreads();

  for (int li = 0; li < L_; ++li) {
    const unsigned short* W = Wl_bf + li * H_ * H_;
#pragma unroll
    for (int t = 0; t < 8; ++t) acc[t] = vzero8();
    for (int k0 = 0; k0 < H_; k0 += 32) {
      v16bf a = frag_rm(&A[0][0], H_, am, k0, lane);
#pragma unroll
      for (int t = 0; t < 8; ++t) {
        v16bf b = frag_rm(W, H_, t * 16 + am, k0, lane);
        acc[t] = wmma_bf(a, b, acc[t]);
      }
    }
    float vals[8][8];
    float sr[8], sq[8];
#pragma unroll
    for (int r = 0; r < 8; ++r) { sr[r] = 0.f; sq[r] = 0.f; }
#pragma unroll
    for (int t = 0; t < 8; ++t) {
      int col = t * 16 + am;
      float bias = b_l[li * H_ + col];
#pragma unroll
      for (int r = 0; r < 8; ++r) {
        float v = fmaxf(acc[t][r] + bias, 0.f);
        vals[t][r] = v; sr[r] += v; sq[r] += v * v;
      }
    }
#pragma unroll
    for (int r = 0; r < 8; ++r) {
      for (int off = 1; off < 16; off <<= 1) {
        sr[r] += __shfl_xor(sr[r], off, 32);
        sq[r] += __shfl_xor(sq[r], off, 32);
      }
    }
#pragma unroll
    for (int t = 0; t < 8; ++t) {
      int col = t * 16 + am;
      float g = g_l[li * H_ + col], be = be_l[li * H_ + col];
#pragma unroll
      for (int r = 0; r < 8; ++r) {
        float mean = sr[r] * (1.f / 128.f);
        float var  = sq[r] * (1.f / 128.f) - mean * mean;
        float rstd = rsqrtf(var + 1e-5f);
        float nv = (vals[t][r] - mean) * rstd * g + be;
        vals[t][r] = nv + bf2f(A[rb + r][col]);
      }
    }
    __syncthreads();
#pragma unroll
    for (int t = 0; t < 8; ++t) {
      int col = t * 16 + am;
#pragma unroll
      for (int r = 0; r < 8; ++r) A[rb + r][col] = f2bf(vals[t][r]);
    }
    __syncthreads();
  }

  for (int j = lane; j < 16 * H_; j += 32) {
    int r = j >> 7, c = j & 127;
    h_bf[(size_t)(r0 + r) * H_ + c] = A[r][c];
  }
}

// ---------------------------------------------------------------------------
// Kernel 2: persistent LSTM + memory-attention scan (see analysis).
// ---------------------------------------------------------------------------
#define SCAN_SMEM_BYTES (131072 + 5*4096 + 16384 + 19552*4)

__global__ __launch_bounds__(256)
void k_scan(const unsigned short* __restrict__ h_bf,
            const unsigned short* __restrict__ Wih, const unsigned short* __restrict__ Whh,
            const unsigned short* __restrict__ Wrh, const unsigned short* __restrict__ Wwh,
            const unsigned short* __restrict__ Weh, const unsigned short* __restrict__ Wqkv,
            const unsigned short* __restrict__ Wo,
            const float* __restrict__ b_ih, const float* __restrict__ b_hh,
            const float* __restrict__ b_rh, const float* __restrict__ b_wh,
            const float* __restrict__ b_eh, const float* __restrict__ b_qkv,
            const float* __restrict__ b_wo,
            const float* __restrict__ W_rg, const float* __restrict__ b_rg,
            const float* __restrict__ W_wg, const float* __restrict__ b_wg,
            const float* __restrict__ mem0,
            float* __restrict__ wm_f32, unsigned short* __restrict__ wm_bf) {
  extern __shared__ char smem[];
  unsigned short* mem_bf = (unsigned short*)smem;          // [16][32][128]
  unsigned short* xt   = mem_bf + B_ * M_ * H_;            // [16][128]
  unsigned short* hsb  = xt  + 16 * H_;
  unsigned short* rkb  = hsb + 16 * H_;
  unsigned short* qpb  = rkb + 16 * H_;
  unsigned short* opb  = qpb + 16 * H_;
  unsigned short* ctxb = opb + 16 * H_;                    // [4][16][128]
  float* qke    = (float*)(ctxb + 4 * 16 * H_);            // [4][16][128]
  float* scores = qke + 4 * 16 * H_;                       // [16][4][32]
  float* aw     = scores + 2048;
  float* qb     = aw + 2048;                               // [16][4]
  float* rg     = qb + 64;
  float* wg     = rg + 16;
  float* attsc  = wg + 16;                                 // [16][32]
  float* attw   = attsc + 512;
  float* wk     = attw + 512;                              // [16][128]
  float* wv     = wk + 2048;
  float* ev     = wv + 2048;

  int tid  = threadIdx.x;
  int lane = tid & 31;
  int w    = tid >> 5;
  int am   = lane & 15;
  int rb   = (lane & 16) ? 8 : 0;
  int col  = w * 16 + am;

  for (int j = tid; j < B_ * M_ * H_; j += 256)
    mem_bf[j] = f2bf(mem0[j & (M_ * H_ - 1)]);
  for (int j = tid; j < 16 * H_; j += 256) hsb[j] = 0;
  __syncthreads();

  v8f cs = vzero8();
  const float bI = b_ih[0 * H_ + col] + b_hh[0 * H_ + col];
  const float bF = b_ih[1 * H_ + col] + b_hh[1 * H_ + col];
  const float bG = b_ih[2 * H_ + col] + b_hh[2 * H_ + col];
  const float bO = b_ih[3 * H_ + col] + b_hh[3 * H_ + col];
  const float scl = 0.17677669529663688f;

  for (int t = 0; t < S_; ++t) {
    for (int j = tid; j < 16 * H_; j += 256) {
      int b = j >> 7, c = j & 127;
      xt[j] = h_bf[((size_t)b * S_ + t) * H_ + c];
    }
    __syncthreads();

    v8f ga[4];
#pragma unroll
    for (int g = 0; g < 4; ++g) ga[g] = vzero8();
    for (int k0 = 0; k0 < H_; k0 += 32) {
      v16bf ax = frag_rm(xt,  H_, am, k0, lane);
      v16bf ah = frag_rm(hsb, H_, am, k0, lane);
#pragma unroll
      for (int g = 0; g < 4; ++g) {
        int n = g * H_ + col;
        ga[g] = wmma_bf(ax, frag_rm(Wih, H_, n, k0, lane), ga[g]);
        ga[g] = wmma_bf(ah, frag_rm(Whh, H_, n, k0, lane), ga[g]);
      }
    }
#pragma unroll
    for (int r = 0; r < 8; ++r) {
      float iv = sigm(ga[0][r] + bI);
      float fv = sigm(ga[1][r] + bF);
      float gv = tanhf(ga[2][r] + bG);
      float ov = sigm(ga[3][r] + bO);
      float c2 = fv * cs[r] + iv * gv;
      cs[r] = c2;
      hsb[(rb + r) * H_ + col] = f2bf(ov * tanhf(c2));
    }
    __syncthreads();

    {
      v8f c = vzero8();
      for (int k0 = 0; k0 < H_; k0 += 32)
        c = wmma_bf(frag_rm(hsb, H_, am, k0, lane),
                    frag_rm(Wrh, H_, col, k0, lane), c);
      float bias = b_rh[col];
#pragma unroll
      for (int r = 0; r < 8; ++r) rkb[(rb + r) * H_ + col] = f2bf(c[r] + bias);
    }
    __syncthreads();

    {
      v8f c = vzero8();
      for (int k0 = 0; k0 < H_; k0 += 32)
        c = wmma_bf(frag_rm(rkb, H_, am, k0, lane),
                    frag_rm(Wqkv, H_, col, k0, lane), c);
      float bias = b_qkv[col];
#pragma unroll
      for (int r = 0; r < 8; ++r) qpb[(rb + r) * H_ + col] = f2bf(c[r] + bias);
    }
    __syncthreads();

    {
      int hh = w >> 1;
      v16bf aq = frag_rm(qpb, H_, am, hh * 32, lane);
      const unsigned short* WkBase = Wqkv + (size_t)(H_ + hh * 32) * H_;
#pragma unroll
      for (int j = 0; j < 4; ++j) {
        int n = ((w & 1) * 4 + j) * 16 + am;
        v8f c = vzero8();
        c = wmma_bf(aq, frag_km(WkBase, H_, n, 0, lane), c);
#pragma unroll
        for (int r = 0; r < 8; ++r) qke[hh * 16 * H_ + (rb + r) * H_ + n] = c[r];
      }
    }
    __syncthreads();

    if (tid < 64) {
      int b = tid >> 2, hh = tid & 3;
      float s = 0.f;
      for (int d = 0; d < 32; ++d)
        s += bf2f(qpb[b * H_ + hh * 32 + d]) * b_qkv[H_ + hh * 32 + d];
      qb[b * 4 + hh] = s;
    } else if (tid < 80) {
      int b = tid - 64; float s = b_rg[0];
      for (int c = 0; c < H_; ++c) s += bf2f(hsb[b * H_ + c]) * W_rg[c];
      rg[b] = sigm(s);
    } else if (tid < 96) {
      int b = tid - 80; float s = b_wg[0];
      for (int c = 0; c < H_; ++c) s += bf2f(hsb[b * H_ + c]) * W_wg[c];
      wg[b] = sigm(s);
    }
    __syncthreads();

#pragma unroll 1
    for (int j = 0; j < 8; ++j) {
      int idx = tid * 8 + j;
      int b = idx >> 7, hh = (idx >> 5) & 3, m = idx & 31;
      float s = qb[b * 4 + hh];
      const float* q = &qke[hh * 16 * H_ + b * H_];
      const unsigned short* mr = &mem_bf[(b * M_ + m) * H_];
      for (int e = 0; e < H_; ++e) s += q[e] * bf2f(mr[e]);
      scores[idx] = s * scl;
    }
    __syncthreads();
    if (tid < 64) {
      float* sc = &scores[tid * 32];
      float mx = sc[0];
      for (int m = 1; m < M_; ++m) mx = fmaxf(mx, sc[m]);
      float sum = 0.f;
      for (int m = 0; m < M_; ++m) { float e = __expf(sc[m] - mx); aw[tid*32+m] = e; sum += e; }
      float inv = 1.f / sum;
      for (int m = 0; m < M_; ++m) aw[tid * 32 + m] *= inv;
    }
    __syncthreads();

    {
      int row = tid >> 2;
      int hh = row >> 4, b = row & 15;
      int e0 = (tid & 3) * 32;
      const float* a_ = &aw[(b * 4 + hh) * 32];
      for (int e = e0; e < e0 + 32; ++e) {
        float s = 0.f;
        for (int m = 0; m < M_; ++m) s += a_[m] * bf2f(mem_bf[(b * M_ + m) * H_ + e]);
        ctxb[row * H_ + e] = f2bf(s);
      }
    }
    __syncthreads();

    {
      int hh = w >> 1, dt = w & 1;
      int colo = hh * 32 + dt * 16 + am;
      v8f c = vzero8();
      for (int k0 = 0; k0 < H_; k0 += 32)
        c = wmma_bf(frag_rm(&ctxb[hh * 16 * H_], H_, am, k0, lane),
                    frag_rm(Wqkv, H_, 2 * H_ + colo, k0, lane), c);
      float bias = b_qkv[2 * H_ + colo];
#pragma unroll
      for (int r = 0; r < 8; ++r) opb[(rb + r) * H_ + colo] = f2bf(c[r] + bias);
    }
    __syncthreads();

    {
      v8f c = vzero8();
      for (int k0 = 0; k0 < H_; k0 += 32)
        c = wmma_bf(frag_rm(opb, H_, am, k0, lane),
                    frag_rm(Wo, H_, col, k0, lane), c);
      float bias = b_wo[col];
#pragma unroll
      for (int r = 0; r < 8; ++r) {
        int b = rb + r;
        float outv = bf2f(xt[b * H_ + col]) + rg[b] * (c[r] + bias);
        size_t o = ((size_t)b * S_ + t) * H_ + col;
        wm_f32[o] = outv;
        wm_bf[o]  = f2bf(outv);
      }
    }
    __syncthreads();

#pragma unroll 1
    for (int j = 0; j < 3; ++j) {
      int tI = w * 3 + j;
      v8f c = vzero8();
      if (tI < 16) {
        int n = tI * 16 + am;
        for (int k0 = 0; k0 < H_; k0 += 32)
          c = wmma_bf(frag_rm(hsb, H_, am, k0, lane),
                      frag_rm(Wwh, H_, n, k0, lane), c);
        float bi = b_wh[n];
        if (tI < 8) {
#pragma unroll
          for (int r = 0; r < 8; ++r) wk[(rb + r) * H_ + n] = c[r] + bi;
        } else {
#pragma unroll
          for (int r = 0; r < 8; ++r) wv[(rb + r) * H_ + (n - H_)] = c[r] + bi;
        }
      } else {
        int n = (tI - 16) * 16 + am;
        for (int k0 = 0; k0 < H_; k0 += 32)
          c = wmma_bf(frag_rm(hsb, H_, am, k0, lane),
                      frag_rm(Weh, H_, n, k0, lane), c);
        float bi = b_eh[n];
#pragma unroll
        for (int r = 0; r < 8; ++r) ev[(rb + r) * H_ + n] = sigm(c[r] + bi);
      }
    }
    __syncthreads();

#pragma unroll 1
    for (int j = 0; j < 2; ++j) {
      int idx = tid * 2 + j;
      int b = idx >> 5, m = idx & 31;
      float s = 0.f;
      const float* wkb = &wk[b * H_];
      for (int e = 0; e < H_; ++e) s += wkb[e] * bf2f(mem_bf[(b * M_ + m) * H_ + e]);
      attsc[idx] = s;
    }
    __syncthreads();
    if (tid < 16) {
      float* sc = &attsc[tid * 32];
      float mx = sc[0];
      for (int m = 1; m < M_; ++m) mx = fmaxf(mx, sc[m]);
      float sum = 0.f;
      for (int m = 0; m < M_; ++m) { float e = __expf(sc[m] - mx); attw[tid*32+m] = e; sum += e; }
      float inv = 1.f / sum;
      for (int m = 0; m < M_; ++m) attw[tid * 32 + m] *= inv;
    }
    __syncthreads();
    for (int j = tid; j < B_ * M_ * H_; j += 256) {
      int b = j >> 12, m = (j >> 7) & 31, e = j & 127;
      float ew = attw[b * 32 + m];
      float mv = bf2f(mem_bf[j]);
      float nv = mv * (1.f - ew * ev[b * H_ + e]) + wg[b] * (ew * wv[b * H_ + e]);
      mem_bf[j] = f2bf(nv);
    }
    __syncthreads();
  }
}

// ---------------------------------------------------------------------------
__global__ __launch_bounds__(256)
void k_qkv(const unsigned short* __restrict__ wm_bf,
           const unsigned short* __restrict__ Wq,
           const float* __restrict__ bias, unsigned short* __restrict__ qkv) {
  int lane = threadIdx.x & 31, w = threadIdx.x >> 5;
  int r0 = blockIdx.x * 128 + w * 16;
  int am = lane & 15, rb = (lane & 16) ? 8 : 0;
  __builtin_prefetch(Wq, 0, 1);
  v16bf afr[4];
#pragma unroll
  for (int kk = 0; kk < 4; ++kk)
    afr[kk] = frag_rm(wm_bf, H_, r0 + am, kk * 32, lane);
#pragma unroll 1
  for (int tO = 0; tO < 24; ++tO) {
    int n = tO * 16 + am;
    v8f c = vzero8();
#pragma unroll
    for (int kk = 0; kk < 4; ++kk)
      c = wmma_bf(afr[kk], frag_rm(Wq, H_, n, kk * 32, lane), c);
    float bi = bias[n];
    float sc = (n < H_) ? 0.25f : 1.f;
#pragma unroll
    for (int r = 0; r < 8; ++r)
      qkv[(size_t)(r0 + rb + r) * H3_ + n] = f2bf((c[r] + bi) * sc);
  }
}

// ---------------------------------------------------------------------------
#define ATTN_PERW (16*512*4 + 16*512*2 + 16*32*2 + 16*32*2)

__global__ __launch_bounds__(128)
void k_attn(const unsigned short* __restrict__ qkv,
            unsigned short* __restrict__ attn_bf) {
  extern __shared__ char sm[];
  int lane = threadIdx.x & 31, w = threadIdx.x >> 5;
  char* base = sm + (size_t)w * ATTN_PERW;
  float* sc          = (float*)base;
  unsigned short* P  = (unsigned short*)(sc + 16 * 512);
  unsigned short* qs = P + 16 * 512;
  unsigned short* kt = qs + 16 * 32;
  int gid = blockIdx.x * 4 + w;
  int qt = gid & 31, h = (gid >> 5) & 7, b = gid >> 8;
  int q0 = qt * 16;
  int am = lane & 15, rb = (lane & 16) ? 8 : 0;

  for (int j = lane; j < 16 * 32; j += 32) {
    int r = j >> 5, d = j & 31;
    qs[j] = (d < 16) ? qkv[((size_t)(b * S_ + q0 + r)) * H3_ + h * 16 + d]
                     : (unsigned short)0;
  }
  __syncthreads();
  v16bf aq = frag_rm(qs, 32, am, 0, lane);

  for (int ktile = 0; ktile < 32; ++ktile) {
    for (int j = lane; j < 16 * 32; j += 32) {
      int r = j >> 5, d = j & 31;
      kt[j] = (d < 16) ? qkv[((size_t)(b * S_ + ktile * 16 + r)) * H3_ + H_ + h * 16 + d]
                       : (unsigned short)0;
    }
    __syncthreads();
    v8f c = vzero8();
    c = wmma_bf(aq, frag_rm(kt, 32, am, 0, lane), c);
    int n = ktile * 16 + am;
#pragma unroll
    for (int r = 0; r < 8; ++r) sc[(rb + r) * 512 + n] = c[r];
    __syncthreads();
  }

  for (int r = 0; r < 16; ++r) {
    float mx = -1e30f;
    for (int c0 = lane; c0 < 512; c0 += 32) mx = fmaxf(mx, sc[r * 512 + c0]);
    for (int off = 16; off; off >>= 1) mx = fmaxf(mx, __shfl_xor(mx, off, 32));
    float sum = 0.f;
    for (int c0 = lane; c0 < 512; c0 += 32) {
      float e = __expf(sc[r * 512 + c0] - mx);
      sum += e;
      P[r * 512 + c0] = f2bf(e);
    }
    for (int off = 16; off; off >>= 1) sum += __shfl_xor(sum, off, 32);
    if (lane == 0) sc[r] = sum;
  }
  __syncthreads();

  v8f oc = vzero8();
  const unsigned short* vbase = qkv + (size_t)b * S_ * H3_ + 2 * H_ + h * 16;
  for (int ks = 0; ks < 16; ++ks) {
    v16bf ap = frag_rm(P, 512, am, ks * 32, lane);
    v16bf bv = frag_km(vbase, H3_, am, ks * 32, lane);
    oc = wmma_bf(ap, bv, oc);
  }
#pragma unroll
  for (int r = 0; r < 8; ++r) {
    int rr = rb + r;
    float inv = 1.f / sc[rr];
    attn_bf[((size_t)(b * S_ + q0 + rr)) * H_ + h * 16 + am] = f2bf(oc[r] * inv);
  }
}

// ---------------------------------------------------------------------------
__global__ __launch_bounds__(256)
void k_proj128(const unsigned short* __restrict__ A,
               const unsigned short* __restrict__ W,
               const float* __restrict__ bias, float* __restrict__ out) {
  int lane = threadIdx.x & 31, w = threadIdx.x >> 5;
  int r0 = blockIdx.x * 128 + w * 16;
  int am = lane & 15, rb = (lane & 16) ? 8 : 0;
  v16bf afr[4];
#pragma unroll
  for (int kk = 0; kk < 4; ++kk) afr[kk] = frag_rm(A, H_, r0 + am, kk * 32, lane);
#pragma unroll 1
  for (int tO = 0; tO < 8; ++tO) {
    int n = tO * 16 + am;
    v8f c = vzero8();
#pragma unroll
    for (int kk = 0; kk < 4; ++kk)
      c = wmma_bf(afr[kk], frag_rm(W, H_, n, kk * 32, lane), c);
    float bi = bias[n];
#pragma unroll
    for (int r = 0; r < 8; ++r) out[(size_t)(r0 + rb + r) * H_ + n] = c[r] + bi;
  }
}

// ---------------------------------------------------------------------------
__global__ __launch_bounds__(256)
void k_co(const float* __restrict__ attn, const float* __restrict__ wm,
          const float* __restrict__ W_ai, const float* __restrict__ b_ai,
          const float* __restrict__ g_ca, const float* __restrict__ be_ca,
          unsigned short* __restrict__ co_bf) {
  int lane = threadIdx.x & 31, w = threadIdx.x >> 5;
  size_t row = (size_t)blockIdx.x * 8 + w;
  const float* av = &attn[row * H_];
  const float* wr = &wm[row * H_];
  float dot = 0.f;
#pragma unroll
  for (int j = 0; j < 4; ++j) { int c = j * 32 + lane; dot += wr[c] * W_ai[c]; }
  for (int off = 16; off; off >>= 1) dot += __shfl_xor(dot, off, 32);
  float inten = sigm(dot + b_ai[0]);
  float vals[4], s = 0.f, s2 = 0.f;
#pragma unroll
  for (int j = 0; j < 4; ++j) {
    int c = j * 32 + lane;
    float v = av[c] * inten + wr[c];
    vals[j] = v; s += v; s2 += v * v;
  }
  for (int off = 16; off; off >>= 1) { s += __shfl_xor(s, off, 32); s2 += __shfl_xor(s2, off, 32); }
  float mean = s * (1.f / 128.f);
  float var  = s2 * (1.f / 128.f) - mean * mean;
  float rstd = rsqrtf(var + 1e-5f);
#pragma unroll
  for (int j = 0; j < 4; ++j) {
    int c = j * 32 + lane;
    co_bf[row * H_ + c] = f2bf((vals[j] - mean) * rstd * g_ca[c] + be_ca[c]);
  }
}

// ---------------------------------------------------------------------------
__global__ __launch_bounds__(256)
void k_out(const unsigned short* __restrict__ co_bf,
           const unsigned short* __restrict__ Wout_bf,
           const float* __restrict__ b_out, float* __restrict__ out) {
  int lane = threadIdx.x & 31, w = threadIdx.x >> 5;
  int r0 = blockIdx.x * 128 + w * 16;
  int am = lane & 15, rb = (lane & 16) ? 8 : 0;
  v16bf afr[4];
#pragma unroll
  for (int kk = 0; kk < 4; ++kk)
    afr[kk] = frag_rm(co_bf, H_, r0 + am, kk * 32, lane);
#pragma unroll 1
  for (int tO = 0; tO < 4; ++tO) {
    int n = tO * 16 + am;
    v8f c = vzero8();
#pragma unroll
    for (int kk = 0; kk < 4; ++kk)
      c = wmma_bf(afr[kk], frag_rm(Wout_bf, H_, n, kk * 32, lane), c);
    float bi = b_out[n];
#pragma unroll
    for (int r = 0; r < 8; ++r) out[(size_t)(r0 + rb + r) * DO_ + n] = c[r] + bi;
  }
}

// ---------------------------------------------------------------------------
extern "C" void kernel_launch(void* const* d_in, const int* in_sizes, int n_in,
                              void* d_out, int out_size, void* d_ws, size_t ws_size,
                              hipStream_t stream) {
  const float* x        = (const float*)d_in[0];
  const float* W_in     = (const float*)d_in[1];
  const float* b_in     = (const float*)d_in[2];
  const float* W_l      = (const float*)d_in[3];
  const float* b_l      = (const float*)d_in[4];
  const float* g_l      = (const float*)d_in[5];
  const float* be_l     = (const float*)d_in[6];
  const float* W_ih     = (const float*)d_in[7];
  const float* W_hh     = (const float*)d_in[8];
  const float* b_ih     = (const float*)d_in[9];
  const float* b_hh     = (const float*)d_in[10];
  const float* mem0     = (const float*)d_in[11];
  const float* W_rh     = (const float*)d_in[12];
  const float* b_rh     = (const float*)d_in[13];
  const float* W_wh     = (const float*)d_in[14];
  const float* b_wh     = (const float*)d_in[15];
  const float* W_eh     = (const float*)d_in[16];
  const float* b_eh     = (const float*)d_in[17];
  const float* W_rg     = (const float*)d_in[18];
  const float* b_rg     = (const float*)d_in[19];
  const float* W_wg     = (const float*)d_in[20];
  const float* b_wg     = (const float*)d_in[21];
  const float* W_wm_qkv = (const float*)d_in[22];
  const float* b_wm_qkv = (const float*)d_in[23];
  const float* W_wm_o   = (const float*)d_in[24];
  const float* b_wm_o   = (const float*)d_in[25];
  const float* W_ca_qkv = (const float*)d_in[26];
  const float* b_ca_qkv = (const float*)d_in[27];
  const float* W_ca_o   = (const float*)d_in[28];
  const float* b_ca_o   = (const float*)d_in[29];
  const float* W_ai     = (const float*)d_in[30];
  const float* b_ai     = (const float*)d_in[31];
  const float* g_ca     = (const float*)d_in[32];
  const float* be_ca    = (const float*)d_in[33];
  const float* W_out    = (const float*)d_in[34];
  const float* b_out    = (const float*)d_in[35];
  (void)in_sizes; (void)n_in; (void)out_size; (void)ws_size;

  char* ws = (char*)d_ws;
  size_t off = 0;
  auto aus = [&](size_t nel) -> unsigned short* {
    unsigned short* p = (unsigned short*)(ws + off);
    off += ((nel * 2 + 255) / 256) * 256; return p;
  };
  auto af = [&](size_t nel) -> float* {
    float* p = (float*)(ws + off);
    off += ((nel * 4 + 255) / 256) * 256; return p;
  };

  unsigned short* Win_bf  = aus(H_ * DI_);
  unsigned short* Wl_bf   = aus(L_ * H_ * H_);
  unsigned short* Wih_bf  = aus(4 * H_ * H_);
  unsigned short* Whh_bf  = aus(4 * H_ * H_);
  unsigned short* Wrh_bf  = aus(H_ * H_);
  unsigned short* Wwh_bf  = aus(2 * H_ * H_);
  unsigned short* Weh_bf  = aus(H_ * H_);
  unsigned short* Wwm_bf  = aus(3 * H_ * H_);
  unsigned short* Wwmo_bf = aus(H_ * H_);
  unsigned short* Wca_bf  = aus(3 * H_ * H_);
  unsigned short* Wcao_bf = aus(H_ * H_);
  unsigned short* Wout_bf = aus(DO_ * H_);
  unsigned short* h_bf    = aus((size_t)BS_ * H_);
  float*          wm_f32  = af((size_t)BS_ * H_);
  unsigned short* wm_bf   = aus((size_t)BS_ * H_);
  unsigned short* qkv_bf  = aus((size_t)BS_ * H3_);
  unsigned short* attn_bf = aus((size_t)BS_ * H_);
  float*          attn_pj = af((size_t)BS_ * H_);
  unsigned short* co_bf   = aus((size_t)BS_ * H_);

  auto conv = [&](const float* s, unsigned short* d, int n) {
    k_f2bf<<<(n + 255) / 256, 256, 0, stream>>>(s, d, n);
  };
  conv(W_in,     Win_bf,  H_ * DI_);
  conv(W_l,      Wl_bf,   L_ * H_ * H_);
  conv(W_ih,     Wih_bf,  4 * H_ * H_);
  conv(W_hh,     Whh_bf,  4 * H_ * H_);
  conv(W_rh,     Wrh_bf,  H_ * H_);
  conv(W_wh,     Wwh_bf,  2 * H_ * H_);
  conv(W_eh,     Weh_bf,  H_ * H_);
  conv(W_wm_qkv, Wwm_bf,  3 * H_ * H_);
  conv(W_wm_o,   Wwmo_bf, H_ * H_);
  conv(W_ca_qkv, Wca_bf,  3 * H_ * H_);
  conv(W_ca_o,   Wcao_bf, H_ * H_);
  conv(W_out,    Wout_bf, DO_ * H_);

  k_mlp<<<BS_ / 128, 256, 0, stream>>>(x, Win_bf, b_in, Wl_bf, b_l, g_l, be_l, h_bf);

  k_scan<<<1, 256, SCAN_SMEM_BYTES, stream>>>(
      h_bf, Wih_bf, Whh_bf, Wrh_bf, Wwh_bf, Weh_bf, Wwm_bf, Wwmo_bf,
      b_ih, b_hh, b_rh, b_wh, b_eh, b_wm_qkv, b_wm_o,
      W_rg, b_rg, W_wg, b_wg, mem0, wm_f32, wm_bf);

  k_qkv<<<BS_ / 128, 256, 0, stream>>>(wm_bf, Wca_bf, b_ca_qkv, qkv_bf);

  k_attn<<<(B_ * 8 * 32) / 4, 128, 4 * ATTN_PERW, stream>>>(qkv_bf, attn_bf);

  k_proj128<<<BS_ / 128, 256, 0, stream>>>(attn_bf, Wcao_bf, b_ca_o, attn_pj);

  k_co<<<BS_ / 8, 256, 0, stream>>>(attn_pj, wm_f32, W_ai, b_ai, g_ca, be_ca, co_bf);

  k_out<<<BS_ / 128, 256, 0, stream>>>(co_bf, Wout_bf, b_out, (float*)d_out);
}
